// FourierTrunkNetwork_40733469835578
// MI455X (gfx1250) — matrix operand order repaired and checked by reading, not compiled, not run on hardware
//
#include <hip/hip_runtime.h>
#include <hip/hip_bf16.h>
#include <math.h>

typedef __attribute__((ext_vector_type(16))) _Float16 v16h;
typedef __attribute__((ext_vector_type(8)))  _Float16 v8h;
typedef __attribute__((ext_vector_type(8)))  float    v8f;

#define SDIM   3
#define FMODES 64
#define IN_DIM 387          // 3 + 3*64*2
#define KPAD1  448          // 387 padded to multiple of 64 (clean 2-phase pipeline)
#define H1D    512
#define H2D    1024
#define LD     256
#define NROWS  128          // rows per workgroup (8 row-tiles of 16)
#define NRT    8            // row tiles per block
#define CHUNK  128          // h2 columns produced+consumed per fusion step
#define NCHUNK (H2D / CHUNK)
#define EPSBN  1e-5f

// ---------------------------------------------------------------------------
// helpers
// ---------------------------------------------------------------------------
__device__ __forceinline__ float gelu_exact(float x) {
  return 0.5f * x * (1.0f + erff(x * 0.70710678118654752f));
}

// A fragment (16x32 f16, one row-tile) from LDS.
// lane L<16 holds row M=L, K = kb+{0..15};
// lane L>=16 holds row M=L-16, K = kb+{16..31} (matches B fragment order).
__device__ __forceinline__ v16h load_a_frag(const _Float16* __restrict__ base,
                                            int stride, int lane) {
  const int r  = lane & 15;
  const int hi = lane >> 4;               // 0 or 1
  const _Float16* p = base + r * stride + hi * 16;
  union { v16h v; v8h h[2]; } u;
  u.h[0] = *(const v8h*)(p);
  u.h[1] = *(const v8h*)(p + 8);
  return u.v;
}

// B fragment (32x16 f16): column n of B == row n of W (row-major, contiguous K).
// lane L<16 -> col n0+L, K=kb+0..15 ; lane L>=16 -> col n0+L-16, K=kb+16..31.
__device__ __forceinline__ v16h load_b_frag(const _Float16* __restrict__ W,
                                            int strideK, int n0, int kb, int lane) {
  const int n = n0 + (lane & 15);
  const int k = kb + ((lane >> 4) << 4);
  return *(const v16h*)(W + (size_t)n * (size_t)strideK + k);
}

// One column-tile x RT row-tiles, accumulating into acc[].
// Explicit 2-deep software pipeline: while WMMAs for one 32-wide half-step
// execute, the other fragment buffer's DS/VMEM loads are in flight.
// Requires K % 64 == 0 and K >= 64 (all call sites: 448 / 512 / 128).
template <int RT>
__device__ __forceinline__ void gemm_acc(const _Float16* __restrict__ A, int sa,
                                         const _Float16* __restrict__ W, int sk,
                                         int n0, int K, int lane, v8f* acc) {
  v16h xa[RT], xb[RT], ba, bb;
  ba = load_b_frag(W, sk, n0, 0, lane);
#pragma unroll
  for (int rt = 0; rt < RT; ++rt) xa[rt] = load_a_frag(A + rt * 16 * sa, sa, lane);

  int kb = 0;
  for (; kb < K - 64; kb += 64) {
    bb = load_b_frag(W, sk, n0, kb + 32, lane);
#pragma unroll
    for (int rt = 0; rt < RT; ++rt)
      xb[rt] = load_a_frag(A + rt * 16 * sa + kb + 32, sa, lane);
#pragma unroll
    for (int rt = 0; rt < RT; ++rt)
      acc[rt] = __builtin_amdgcn_wmma_f32_16x16x32_f16(false, xa[rt], false, ba,
                                                       (short)0, acc[rt], false, false);
    ba = load_b_frag(W, sk, n0, kb + 64, lane);
#pragma unroll
    for (int rt = 0; rt < RT; ++rt)
      xa[rt] = load_a_frag(A + rt * 16 * sa + kb + 64, sa, lane);
#pragma unroll
    for (int rt = 0; rt < RT; ++rt)
      acc[rt] = __builtin_amdgcn_wmma_f32_16x16x32_f16(false, xb[rt], false, bb,
                                                       (short)0, acc[rt], false, false);
  }
  // tail: kb == K-64, no further prefetch
  bb = load_b_frag(W, sk, n0, kb + 32, lane);
#pragma unroll
  for (int rt = 0; rt < RT; ++rt)
    xb[rt] = load_a_frag(A + rt * 16 * sa + kb + 32, sa, lane);
#pragma unroll
  for (int rt = 0; rt < RT; ++rt)
    acc[rt] = __builtin_amdgcn_wmma_f32_16x16x32_f16(false, xa[rt], false, ba,
                                                     (short)0, acc[rt], false, false);
#pragma unroll
  for (int rt = 0; rt < RT; ++rt)
    acc[rt] = __builtin_amdgcn_wmma_f32_16x16x32_f16(false, xb[rt], false, bb,
                                                     (short)0, acc[rt], false, false);
}

// C/D layout: VGPR r, lanes 0-15 -> M=r, N=lane; lanes 16-31 -> M=r+8, N=lane-16.
__device__ __forceinline__ void store_tile_gelu_lds(_Float16* __restrict__ H, int so,
                                                    int rowbase, int n0, int lane,
                                                    float cn, v8f acc) {
  const int col = n0 + (lane & 15);
  const int mro = rowbase + ((lane >> 4) << 3);
#pragma unroll
  for (int r = 0; r < 8; ++r)
    H[(mro + r) * so + col] = (_Float16)gelu_exact(acc[r] + cn);
}

// ---------------------------------------------------------------------------
// prep kernels: fold BN into weights, convert f32 -> f16, zero-pad K
// ---------------------------------------------------------------------------
__global__ void prep_w_bn(const float* __restrict__ W, const float* __restrict__ b,
                          const float* __restrict__ g, const float* __restrict__ be,
                          const float* __restrict__ mu, const float* __restrict__ var,
                          _Float16* __restrict__ Wh, float* __restrict__ c,
                          int H, int K, int KP) {
  int i = blockIdx.x * 256 + threadIdx.x;
  if (i >= H * KP) return;
  int h = i / KP;
  int k = i - h * KP;
  float s = g[h] * rsqrtf(var[h] + EPSBN);
  float w = (k < K) ? W[(size_t)h * K + k] * s : 0.0f;
  Wh[i] = (_Float16)w;
  if (k == 0) c[h] = (b[h] - mu[h]) * s + be[h];
}

__global__ void prep_w_plain(const float* __restrict__ W, _Float16* __restrict__ Wh, int total) {
  int i = blockIdx.x * 256 + threadIdx.x;
  if (i < total) Wh[i] = (_Float16)W[i];
}

// ---------------------------------------------------------------------------
// fused trunk, 128 rows/block:
//   feats -> GEMM1+BN+GELU -> { per 128-col chunk: GEMM2+BN+GELU -> GEMM3 k-slice }
// GEMM3 partial sums live in LDS between chunks ([tile][lane][8] f32, b128 io).
// ---------------------------------------------------------------------------
__global__ __launch_bounds__(256)
void trunk_fused(const float* __restrict__ y,        // [N,3]
                 const float* __restrict__ freqs,    // [3,64]
                 const _Float16* __restrict__ W1h,   // [512,448] BN-folded
                 const float* __restrict__ c1,       // [512]
                 const _Float16* __restrict__ W2h,   // [1024,512] BN-folded
                 const float* __restrict__ c2,       // [1024]
                 const _Float16* __restrict__ W3h,   // [256,1024]
                 const float* __restrict__ b3,       // [256]
                 float* __restrict__ out)            // [N,256]
{
  extern __shared__ __align__(16) char smem[];
  _Float16* h1    = (_Float16*)smem;                        // 128*512 f16 = 128KB
  float*    oaccL = (float*)(smem + NROWS * H1D * 2);       // 128 tiles*32 lanes*8 f32 = 128KB
  _Float16* hc    = (_Float16*)(smem + NROWS * H1D * 2 + NROWS * LD * 4); // 32KB
  _Float16* fx    = (_Float16*)oaccL;                        // feats alias (112KB, dead after st.1)

  __shared__ float ys[NROWS * SDIM];
  __shared__ float fs[SDIM * FMODES];

  const int tid  = threadIdx.x;
  const int lane = tid & 31;
  const int wave = tid >> 5;
  const int row0 = blockIdx.x * NROWS;

  for (int i = tid; i < NROWS * SDIM; i += 256) ys[i] = y[(size_t)row0 * SDIM + i];
  for (int i = tid; i < SDIM * FMODES; i += 256) fs[i] = freqs[i];
  __syncthreads();

  // ---- Stage 0: fourier features, f16 into LDS (hardware v_sin/v_cos) ----
  for (int i = tid; i < NROWS * KPAD1; i += 256) {
    int r = i / KPAD1;
    int k = i - r * KPAD1;
    float v = 0.0f;
    if (k < SDIM) {
      v = ys[r * SDIM + k];
    } else if (k < IN_DIM) {
      int j  = k - SDIM;
      int d  = j >> 7;           // / (FMODES*2)
      int rm = j & 127;
      int m  = rm >> 1;
      float ph = ys[r * SDIM + d] * fs[d * FMODES + m];
      v = (rm & 1) ? __cosf(ph) : __sinf(ph);
    }
    fx[r * KPAD1 + k] = (_Float16)v;
  }
  __syncthreads();

  // ---- Stage 1: h1 = gelu(fx @ W1' + c1), K=448, N=512 (32 col tiles) ----
  for (int ct = wave; ct < H1D / 16; ct += 8) {
    const int n0 = ct * 16;
    if (n0 + 128 < H1D)
      __builtin_prefetch((const void*)(W1h + (size_t)(n0 + 128 + (lane & 15)) * KPAD1), 0, 1);
    v8f acc[NRT] = {};
    gemm_acc<NRT>(fx, KPAD1, W1h, KPAD1, n0, KPAD1, lane, acc);
    float cn = c1[n0 + (lane & 15)];
#pragma unroll
    for (int rt = 0; rt < NRT; ++rt)
      store_tile_gelu_lds(h1, H1D, rt * 16, n0, lane, cn, acc[rt]);
  }
  __syncthreads();   // h1 ready; fx dead -> oaccL region free

  // ---- Stage 2+3 fused over 128-column chunks of h2 ----
  for (int c = 0; c < NCHUNK; ++c) {
    // Stage 2: h2 chunk cols [c*128, c*128+128); one 16-col tile per wave.
    {
      const int n0g = c * CHUNK + wave * 16;            // column in full h2
      if (c + 1 < NCHUNK)
        __builtin_prefetch((const void*)(W2h + (size_t)(n0g + CHUNK + (lane & 15)) * H1D), 0, 1);
      v8f acc[NRT] = {};
      gemm_acc<NRT>(h1, H1D, W2h, H1D, n0g, H1D, lane, acc);
      float cn = c2[n0g + (lane & 15)];
      const int n0l = wave * 16;                         // column within chunk
#pragma unroll
      for (int rt = 0; rt < NRT; ++rt)
        store_tile_gelu_lds(hc, CHUNK, rt * 16, n0l, lane, cn, acc[rt]);
    }
    __syncthreads();   // chunk ready for GEMM3

    // Stage 3 partial: out_partial += chunk @ W3[:, c*128:(c+1)*128]^T
    // Partial sums parked in LDS between chunks; each tile owned by one wave.
#pragma unroll
    for (int cc = 0; cc < 2; ++cc) {
      const int n0 = (wave * 2 + cc) * 16;
      v8f acc[NRT];
      float* tbase = oaccL + ((size_t)(wave * 2 + cc) * NRT * 32 + lane) * 8;
      if (c == 0) {
#pragma unroll
        for (int rt = 0; rt < NRT; ++rt) acc[rt] = (v8f){};
      } else {
#pragma unroll
        for (int rt = 0; rt < NRT; ++rt) acc[rt] = *(const v8f*)(tbase + rt * 32 * 8);
      }
      gemm_acc<NRT>(hc, CHUNK, W3h + c * CHUNK, H2D, n0, CHUNK, lane, acc);
      if (c == NCHUNK - 1) {
        // final chunk: fuse bias + global store
        const int col = n0 + (lane & 15);
        const float bn = b3[col];
        const int mro = (lane >> 4) << 3;
#pragma unroll
        for (int rt = 0; rt < NRT; ++rt) {
#pragma unroll
          for (int r = 0; r < 8; ++r)
            out[(size_t)(row0 + rt * 16 + mro + r) * LD + col] = acc[rt][r] + bn;
        }
      } else {
#pragma unroll
        for (int rt = 0; rt < NRT; ++rt) *(v8f*)(tbase + rt * 32 * 8) = acc[rt];
      }
    }
    __syncthreads();   // chunk consumed; safe to overwrite next iteration
  }
}

// ---------------------------------------------------------------------------
extern "C" void kernel_launch(void* const* d_in, const int* in_sizes, int n_in,
                              void* d_out, int out_size, void* d_ws, size_t ws_size,
                              hipStream_t stream) {
  const float* y     = (const float*)d_in[0];
  const float* freqs = (const float*)d_in[1];
  const float* W1 = (const float*)d_in[2];
  const float* b1 = (const float*)d_in[3];
  const float* g1 = (const float*)d_in[4];
  const float* be1 = (const float*)d_in[5];
  const float* mu1 = (const float*)d_in[6];
  const float* var1 = (const float*)d_in[7];
  const float* W2 = (const float*)d_in[8];
  const float* b2 = (const float*)d_in[9];
  const float* g2 = (const float*)d_in[10];
  const float* be2 = (const float*)d_in[11];
  const float* mu2 = (const float*)d_in[12];
  const float* var2 = (const float*)d_in[13];
  const float* W3 = (const float*)d_in[14];
  const float* b3 = (const float*)d_in[15];

  // workspace carve (all offsets 16B aligned)
  char* ws = (char*)d_ws;
  float*    c1  = (float*)ws;                                   // 512 f32
  float*    c2  = (float*)(ws + 2048);                          // 1024 f32
  _Float16* W1h = (_Float16*)(ws + 6144);                       // 512*448 f16
  _Float16* W2h = (_Float16*)(ws + 6144 + (size_t)H1D * KPAD1 * 2);
  _Float16* W3h = (_Float16*)(ws + 6144 + (size_t)H1D * KPAD1 * 2
                                        + (size_t)H2D * H1D * 2);

  prep_w_bn<<<(H1D * KPAD1 + 255) / 256, 256, 0, stream>>>(
      W1, b1, g1, be1, mu1, var1, W1h, c1, H1D, IN_DIM, KPAD1);
  prep_w_bn<<<(H2D * H1D + 255) / 256, 256, 0, stream>>>(
      W2, b2, g2, be2, mu2, var2, W2h, c2, H2D, H1D, H1D);
  prep_w_plain<<<(LD * H2D + 255) / 256, 256, 0, stream>>>(W3, W3h, LD * H2D);

  const int N = in_sizes[0] / SDIM;          // 262144, divisible by 128
  const size_t lds = (size_t)NROWS * H1D * 2           // h1:   128KB
                   + (size_t)NROWS * LD * 4            // oacc: 128KB (feats alias)
                   + (size_t)NROWS * CHUNK * 2;        // h2 chunk: 32KB
  trunk_fused<<<N / NROWS, 256, lds, stream>>>(
      y, freqs, W1h, c1, W2h, c2, W3h, b3, (float*)d_out);
}
